// EndToEndModel_59476707115443
// MI455X (gfx1250) — matrix operand order repaired and checked by reading, not compiled, not run on hardware
//
#include <hip/hip_runtime.h>
#include <hip/hip_bf16.h>
#include <math.h>

// ---------------------------------------------------------------------------
// Types for CDNA5 WMMA (wave32): v_wmma_f32_16x16x32_bf16
// ---------------------------------------------------------------------------
typedef __bf16 bf16;
typedef __attribute__((ext_vector_type(8)))  __bf16 v8bf;
typedef __attribute__((ext_vector_type(16))) __bf16 v16bf;
typedef __attribute__((ext_vector_type(8)))  float  v8f;

#define Dm   768
#define FFm  3072
#define Bm   8
#define Sm   256
#define Mm   2048          // B*S
#define Lm   12
#define Hm   12
#define HDm  64
#define Rm   16
#define Tm   3

// ---------------------------------------------------------------------------
// One K=32 WMMA step. A: M-row per lane (contiguous K layout:
// lane<16 holds K 0..15, lane>=16 holds K 16..31).
// B operand comes from a transposed weight (N x K row-major): lane n holds
// 16 contiguous K values of column n (K offset half*16).
// ---------------------------------------------------------------------------
__device__ __forceinline__ v8f wmma_step(const bf16* __restrict__ Arow,
                                         const bf16* __restrict__ Brow,
                                         int k, int half, v8f acc) {
  v8bf a0 = *(const v8bf*)(Arow + k + half * 16);
  v8bf a1 = *(const v8bf*)(Arow + k + half * 16 + 8);
  v16bf a;
#pragma unroll
  for (int i = 0; i < 8; ++i) { a[i] = a0[i]; a[i + 8] = a1[i]; }
  v16bf b = *(const v16bf*)(Brow + k + half * 16);
  return __builtin_amdgcn_wmma_f32_16x16x32_bf16(false, a, false, b,
                                                 (short)0, acc, false, false);
}

__device__ __forceinline__ float gelu_tanh(float x) {
  float x3 = x * x * x;
  return 0.5f * x * (1.0f + tanhf(0.7978845608028654f * (x + 0.044715f * x3)));
}

__device__ __forceinline__ float block_sum256(float v, float* red) {
  int t = threadIdx.x;
  red[t] = v;
  __syncthreads();
  for (int s = 128; s > 0; s >>= 1) {
    if (t < s) red[t] += red[t + s];
    __syncthreads();
  }
  float r = red[0];
  __syncthreads();
  return r;
}

// ---------------------------------------------------------------------------
// fp32 -> bf16 weight conversion with transpose. src: (L, K, N) f32,
// dst: (L, N, K) bf16  (so dst rows are WMMA B-operand friendly).
// ---------------------------------------------------------------------------
__global__ void cvt_t_kernel(const float* __restrict__ src, bf16* __restrict__ dst,
                             int L, int K, int N) {
  int idx = blockIdx.x * 256 + threadIdx.x;
  int total = L * N * K;
  if (idx >= total) return;
  int l = idx / (N * K);
  int rem = idx - l * (N * K);
  int n = rem / K;
  int k = rem - n * K;
  dst[idx] = (bf16)src[(size_t)l * K * N + (size_t)k * N + n];
}

__global__ void cvt_kernel(const float* __restrict__ src, bf16* __restrict__ dst,
                           int n) {
  int idx = blockIdx.x * 256 + threadIdx.x;
  if (idx < n) dst[idx] = (bf16)src[idx];
}

// ---------------------------------------------------------------------------
// Embedding gather + LayerNorm. One block per token row.
// ---------------------------------------------------------------------------
__global__ void embed_ln_kernel(const int* __restrict__ ids,
                                const float* __restrict__ wemb,
                                const float* __restrict__ pemb,
                                const float* __restrict__ g,
                                const float* __restrict__ be,
                                float* __restrict__ hf, bf16* __restrict__ hb) {
  __shared__ float red[256];
  int row = blockIdx.x;
  int s = row & (Sm - 1);
  int tok = ids[row];
  int t = threadIdx.x;
  float x[3];
#pragma unroll
  for (int i = 0; i < 3; ++i) {
    int c = t + i * 256;
    x[i] = wemb[(size_t)tok * Dm + c] + pemb[(size_t)(s + 2) * Dm + c];
  }
  float mean = block_sum256(x[0] + x[1] + x[2], red) * (1.0f / Dm);
  float sq = x[0] * x[0] + x[1] * x[1] + x[2] * x[2];
  float var = block_sum256(sq, red) * (1.0f / Dm) - mean * mean;
  float rstd = rsqrtf(var + 1e-5f);
#pragma unroll
  for (int i = 0; i < 3; ++i) {
    int c = t + i * 256;
    float y = (x[i] - mean) * rstd * g[c] + be[c];
    hf[(size_t)row * Dm + c] = y;
    hb[(size_t)row * Dm + c] = (bf16)y;
  }
}

// Residual add + LayerNorm. One block per row.
__global__ void add_ln_kernel(const float* __restrict__ a,
                              const float* __restrict__ bsrc,
                              const float* __restrict__ g,
                              const float* __restrict__ be,
                              float* __restrict__ of, bf16* __restrict__ ob) {
  __shared__ float red[256];
  int row = blockIdx.x;
  int t = threadIdx.x;
  float x[3];
#pragma unroll
  for (int i = 0; i < 3; ++i) {
    int c = t + i * 256;
    x[i] = a[(size_t)row * Dm + c] + bsrc[(size_t)row * Dm + c];
  }
  float mean = block_sum256(x[0] + x[1] + x[2], red) * (1.0f / Dm);
  float sq = x[0] * x[0] + x[1] * x[1] + x[2] * x[2];
  float var = block_sum256(sq, red) * (1.0f / Dm) - mean * mean;
  float rstd = rsqrtf(var + 1e-5f);
#pragma unroll
  for (int i = 0; i < 3; ++i) {
    int c = t + i * 256;
    float y = (x[i] - mean) * rstd * g[c] + be[c];
    of[(size_t)row * Dm + c] = y;
    ob[(size_t)row * Dm + c] = (bf16)y;
  }
}

// ---------------------------------------------------------------------------
// Generic bf16 GEMM: C(MxN) = act(A(MxK) * Bt(NxK)^T + bias).
// 8 waves / block; each wave owns one 16x16 tile; waves in a block share the
// same N-tile of B for cache reuse. grid = (N/16, M/128), block = 256.
// act: 0 = none, 1 = gelu(tanh).
// ---------------------------------------------------------------------------
__global__ void gemm_bf16_kernel(const bf16* __restrict__ A,
                                 const bf16* __restrict__ Bt,
                                 const float* __restrict__ bias,
                                 float* __restrict__ Cf, bf16* __restrict__ Cb,
                                 int M, int N, int K, int act) {
  int wave = threadIdx.x >> 5;
  int lane = threadIdx.x & 31;
  int half = lane >> 4;
  int r = lane & 15;
  int mBase = (blockIdx.y * 8 + wave) * 16;
  int nBase = blockIdx.x * 16;

  const bf16* Arow = A + (size_t)(mBase + r) * K;
  const bf16* Brow = Bt + (size_t)(nBase + r) * K;

  v8f acc = {};
  for (int k = 0; k < K; k += 32) {
    __builtin_prefetch(Brow + k + 256, 0, 1);   // global_prefetch_b8
    acc = wmma_step(Arow, Brow, k, half, acc);
  }

#pragma unroll
  for (int j = 0; j < 8; ++j) {
    int row = mBase + j + half * 8;
    int col = nBase + r;
    float v = acc[j] + bias[col];
    if (act == 1) v = gelu_tanh(v);
    if (Cf) Cf[(size_t)row * N + col] = v;
    if (Cb) Cb[(size_t)row * N + col] = (bf16)v;
  }
}

// ---------------------------------------------------------------------------
// LoRA stage 1: tmp(M x 16) = h(M x 768) * A_l(16 x 768)^T, adapter chosen
// per-sample via task[b]. grid = (1, 16), block = 256.
// ---------------------------------------------------------------------------
__global__ void lora_gemm1_kernel(const bf16* __restrict__ h,
                                  const bf16* __restrict__ loraA,
                                  const int* __restrict__ task, int layer,
                                  bf16* __restrict__ tmp) {
  int wave = threadIdx.x >> 5;
  int lane = threadIdx.x & 31;
  int half = lane >> 4;
  int r = lane & 15;
  int mBase = (blockIdx.y * 8 + wave) * 16;
  int b = mBase >> 8;                            // /S
  const bf16* Bt = loraA + ((size_t)(task[b] * Lm + layer) * Rm) * Dm;
  const bf16* Arow = h + (size_t)(mBase + r) * Dm;
  const bf16* Brow = Bt + (size_t)r * Dm;

  v8f acc = {};
  for (int k = 0; k < Dm; k += 32) acc = wmma_step(Arow, Brow, k, half, acc);

#pragma unroll
  for (int j = 0; j < 8; ++j)
    tmp[(size_t)(mBase + j + half * 8) * Rm + r] = (bf16)acc[j];
}

// ---------------------------------------------------------------------------
// LoRA stage 2: C(M x 768) += 2.0 * tmp(M x 16) * B_l(768 x 16)^T.
// K=16 zero-padded to one 16x16x32 WMMA. grid = (48, 16), block = 256.
// ---------------------------------------------------------------------------
__global__ void lora_gemm2_kernel(const bf16* __restrict__ tmp,
                                  const bf16* __restrict__ loraB,
                                  const int* __restrict__ task, int layer,
                                  float* __restrict__ Cf, bf16* __restrict__ Cb) {
  int wave = threadIdx.x >> 5;
  int lane = threadIdx.x & 31;
  int half = lane >> 4;
  int r = lane & 15;
  int mBase = (blockIdx.y * 8 + wave) * 16;
  int nBase = blockIdx.x * 16;
  int b = mBase >> 8;
  const bf16* Bt = loraB + (size_t)(task[b] * Lm + layer) * Dm * Rm;

  const bf16* Ar = tmp + (size_t)(mBase + r) * Rm;
  v16bf a;
#pragma unroll
  for (int i = 0; i < 8; ++i) { a[i] = half ? (bf16)0.0f : Ar[i]; a[i + 8] = half ? (bf16)0.0f : Ar[i + 8]; }
  v16bf bb;
  if (half == 0) {
    const bf16* Br = Bt + (size_t)(nBase + r) * Rm;
#pragma unroll
    for (int i = 0; i < 16; ++i) bb[i] = Br[i];
  } else {
#pragma unroll
    for (int i = 0; i < 16; ++i) bb[i] = (bf16)0.0f;
  }
  v8f acc = {};
  acc = __builtin_amdgcn_wmma_f32_16x16x32_bf16(false, a, false, bb,
                                                (short)0, acc, false, false);
#pragma unroll
  for (int j = 0; j < 8; ++j) {
    int row = mBase + j + half * 8;
    int col = nBase + r;
    float v = Cf[(size_t)row * Dm + col] + 2.0f * acc[j];   // LORA_SCALE = 2
    Cf[(size_t)row * Dm + col] = v;
    Cb[(size_t)row * Dm + col] = (bf16)v;
  }
}

// ---------------------------------------------------------------------------
// Attention: 1 wave per (b, h, 16-query tile). Scores via WMMA into LDS,
// row softmax, P*V via WMMA; row inverse folded into the output store.
// grid = B*H*(S/16) = 1536, block = 32.
// ---------------------------------------------------------------------------
__global__ void attn_kernel(const bf16* __restrict__ qb,
                            const bf16* __restrict__ kb,
                            const bf16* __restrict__ vb,
                            const int* __restrict__ amask,
                            bf16* __restrict__ ob) {
  __shared__ float sc[16 * Sm];
  __shared__ float rinv[16];

  int idx = blockIdx.x;
  int qt = idx & 15;
  int h = (idx >> 4) % Hm;
  int b = idx / (16 * Hm);
  int lane = threadIdx.x;
  int half = lane >> 4;
  int r = lane & 15;

  int rowBase = b * Sm + qt * 16;
  const bf16* Q = qb + (size_t)rowBase * Dm + h * HDm;
  const bf16* Aq = Q + (size_t)r * Dm;

  // scores = Q K^T / 8 + mask_bias
  for (int kt = 0; kt < 16; ++kt) {
    const bf16* Kp = kb + (size_t)(b * Sm + kt * 16) * Dm + h * HDm;
    const bf16* Bk = Kp + (size_t)r * Dm;       // lane n = K row n = col n of K^T
    v8f acc = {};
    for (int kk = 0; kk < HDm; kk += 32) acc = wmma_step(Aq, Bk, kk, half, acc);
#pragma unroll
    for (int j = 0; j < 8; ++j) {
      int m = j + half * 8;
      int col = kt * 16 + r;
      float bias = amask[b * Sm + col] ? 0.0f : -1e9f;
      sc[m * Sm + col] = acc[j] * 0.125f + bias;
    }
  }
  __syncthreads();

  // row softmax (unnormalized exp stored, inverse sum kept per row)
  if (lane < 16) {
    float mx = -3.0e38f;
    for (int c = 0; c < Sm; ++c) mx = fmaxf(mx, sc[lane * Sm + c]);
    float s = 0.0f;
    for (int c = 0; c < Sm; ++c) {
      float e = __expf(sc[lane * Sm + c] - mx);
      sc[lane * Sm + c] = e;
      s += e;
    }
    rinv[lane] = 1.0f / s;
  }
  __syncthreads();

  // O = P * V
  const float* Pr = sc + r * Sm;
  for (int nt = 0; nt < 4; ++nt) {
    v8f acc = {};
    for (int kk = 0; kk < Sm; kk += 32) {
      v16bf a;
#pragma unroll
      for (int i = 0; i < 8; ++i) {
        a[i] = (bf16)Pr[kk + half * 16 + i];
        a[i + 8] = (bf16)Pr[kk + half * 16 + 8 + i];
      }
      v16bf bv;
#pragma unroll
      for (int i = 0; i < 16; ++i)
        bv[i] = vb[(size_t)(b * Sm + kk + half * 16 + i) * Dm + h * HDm + nt * 16 + r];
      acc = __builtin_amdgcn_wmma_f32_16x16x32_bf16(false, a, false, bv,
                                                    (short)0, acc, false, false);
    }
#pragma unroll
    for (int j = 0; j < 8; ++j) {
      int m = j + half * 8;
      ob[(size_t)(rowBase + m) * Dm + h * HDm + nt * 16 + r] =
          (bf16)(acc[j] * rinv[m]);
    }
  }
}

// ---------------------------------------------------------------------------
// Router: per sample, relu(cls @ W1 + b1) @ W2 + b2, /max(temp,.05), argmax.
// grid = 8, block = 256.
// ---------------------------------------------------------------------------
__global__ void router_kernel(const float* __restrict__ h,
                              const float* __restrict__ W1,
                              const float* __restrict__ b1,
                              const float* __restrict__ W2,
                              const float* __restrict__ b2,
                              const float* __restrict__ temp,
                              float* __restrict__ out, int* __restrict__ task) {
  __shared__ float hid[Dm];
  __shared__ float lg[Tm];
  int b = blockIdx.x;
  int t = threadIdx.x;
  const float* cls = h + (size_t)b * Sm * Dm;
#pragma unroll
  for (int i = 0; i < 3; ++i) {
    int col = t + i * 256;
    float s = b1[col];
    for (int k = 0; k < Dm; ++k) s += cls[k] * W1[(size_t)k * Dm + col];
    hid[col] = fmaxf(s, 0.0f);
  }
  __syncthreads();
  if (t < Tm) {
    float s = b2[t];
    for (int k = 0; k < Dm; ++k) s += hid[k] * W2[k * Tm + t];
    s /= fmaxf(temp[0], 0.05f);
    out[b * Tm + t] = s;
    lg[t] = s;
  }
  __syncthreads();
  if (t == 0) {
    int best = 0;
    if (lg[1] > lg[0]) best = 1;
    if (lg[2] > lg[best]) best = 2;
    task[b] = best;
  }
}

// ---------------------------------------------------------------------------
// Classifier head: f = tanh(cls @ cW1[task] + cb1[task]); logits = f @ cW2 + cb2.
// grid = 8, block = 256.
// ---------------------------------------------------------------------------
__global__ void cls_kernel(const float* __restrict__ h,
                           const float* __restrict__ W1,
                           const float* __restrict__ b1,
                           const float* __restrict__ W2,
                           const float* __restrict__ b2,
                           const int* __restrict__ task, float* __restrict__ out) {
  __shared__ float hid[Dm];
  int b = blockIdx.x;
  int t = threadIdx.x;
  int tk = task[b];
  const float* f = h + (size_t)b * Sm * Dm;
#pragma unroll
  for (int i = 0; i < 3; ++i) {
    int col = t + i * 256;
    float s = b1[(size_t)tk * Dm + col];
    for (int k = 0; k < Dm; ++k)
      s += f[k] * W1[((size_t)tk * Dm + k) * Dm + col];
    hid[col] = tanhf(s);
  }
  __syncthreads();
  if (t < 2) {
    float s = b2[tk * 2 + t];
    for (int k = 0; k < Dm; ++k) s += hid[k] * W2[((size_t)tk * Dm + k) * 2 + t];
    out[b * 2 + t] = s;
  }
}

// ---------------------------------------------------------------------------
// Host orchestration
// ---------------------------------------------------------------------------
extern "C" void kernel_launch(void* const* d_in, const int* in_sizes, int n_in,
                              void* d_out, int out_size, void* d_ws, size_t ws_size,
                              hipStream_t stream) {
  (void)in_sizes; (void)n_in; (void)out_size; (void)ws_size;

  const int*   ids    = (const int*)d_in[0];
  const int*   amask  = (const int*)d_in[1];
  const float* wemb   = (const float*)d_in[2];
  const float* pemb   = (const float*)d_in[3];
  const float* emb_g  = (const float*)d_in[4];
  const float* emb_b  = (const float*)d_in[5];
  const float* Wq     = (const float*)d_in[6];
  const float* Wk     = (const float*)d_in[7];
  const float* Wv     = (const float*)d_in[8];
  const float* Wo     = (const float*)d_in[9];
  const float* Wi     = (const float*)d_in[10];
  const float* Wo2    = (const float*)d_in[11];
  const float* bq     = (const float*)d_in[12];
  const float* bk     = (const float*)d_in[13];
  const float* bv     = (const float*)d_in[14];
  const float* bo     = (const float*)d_in[15];
  const float* bi     = (const float*)d_in[16];
  const float* bo2    = (const float*)d_in[17];
  const float* g1     = (const float*)d_in[18];
  const float* b1     = (const float*)d_in[19];
  const float* g2     = (const float*)d_in[20];
  const float* b2     = (const float*)d_in[21];
  const float* rW1    = (const float*)d_in[22];
  const float* rb1    = (const float*)d_in[23];
  const float* rW2    = (const float*)d_in[24];
  const float* rb2    = (const float*)d_in[25];
  const float* temp   = (const float*)d_in[26];
  const float* qA     = (const float*)d_in[27];
  const float* qB     = (const float*)d_in[28];
  const float* vA     = (const float*)d_in[29];
  const float* vB     = (const float*)d_in[30];
  const float* cW1    = (const float*)d_in[31];
  const float* cb1    = (const float*)d_in[32];
  const float* cW2    = (const float*)d_in[33];
  const float* cb2    = (const float*)d_in[34];
  float* out = (float*)d_out;

  // ---- workspace carve-up ----
  char* base = (char*)d_ws;
  size_t off = 0;
  auto alloc = [&](size_t bytes) -> void* {
    void* p = base + off;
    off += (bytes + 255) & ~(size_t)255;
    return p;
  };
  bf16* wqT  = (bf16*)alloc((size_t)Lm * Dm * Dm * 2);
  bf16* wkT  = (bf16*)alloc((size_t)Lm * Dm * Dm * 2);
  bf16* wvT  = (bf16*)alloc((size_t)Lm * Dm * Dm * 2);
  bf16* woT  = (bf16*)alloc((size_t)Lm * Dm * Dm * 2);
  bf16* wiT  = (bf16*)alloc((size_t)Lm * Dm * FFm * 2);
  bf16* wo2T = (bf16*)alloc((size_t)Lm * Dm * FFm * 2);
  bf16* qA_b = (bf16*)alloc((size_t)Tm * Lm * Rm * Dm * 2);
  bf16* qB_b = (bf16*)alloc((size_t)Tm * Lm * Dm * Rm * 2);
  bf16* vA_b = (bf16*)alloc((size_t)Tm * Lm * Rm * Dm * 2);
  bf16* vB_b = (bf16*)alloc((size_t)Tm * Lm * Dm * Rm * 2);
  float* h_f  = (float*)alloc((size_t)Mm * Dm * 4);
  bf16*  h_b  = (bf16*) alloc((size_t)Mm * Dm * 2);
  float* q_f  = (float*)alloc((size_t)Mm * Dm * 4);
  bf16*  q_b  = (bf16*) alloc((size_t)Mm * Dm * 2);
  bf16*  k_b  = (bf16*) alloc((size_t)Mm * Dm * 2);
  float* v_f  = (float*)alloc((size_t)Mm * Dm * 4);
  bf16*  v_b  = (bf16*) alloc((size_t)Mm * Dm * 2);
  bf16*  tmp_b= (bf16*) alloc((size_t)Mm * Rm * 2);
  bf16*  o_b  = (bf16*) alloc((size_t)Mm * Dm * 2);
  float* t1_f = (float*)alloc((size_t)Mm * Dm * 4);
  float* h1_f = (float*)alloc((size_t)Mm * Dm * 4);
  bf16*  h1_b = (bf16*) alloc((size_t)Mm * Dm * 2);
  bf16*  ff1_b= (bf16*) alloc((size_t)Mm * FFm * 2);
  float* ff2_f= (float*)alloc((size_t)Mm * Dm * 4);
  int*   task = (int*)  alloc(Bm * 4);

  auto cblk = [](size_t n) { return (unsigned)((n + 255) / 256); };

  // ---- weight conversion (fp32 -> bf16, transposed) ----
  cvt_t_kernel<<<cblk((size_t)Lm*Dm*Dm), 256, 0, stream>>>(Wq,  wqT,  Lm, Dm,  Dm);
  cvt_t_kernel<<<cblk((size_t)Lm*Dm*Dm), 256, 0, stream>>>(Wk,  wkT,  Lm, Dm,  Dm);
  cvt_t_kernel<<<cblk((size_t)Lm*Dm*Dm), 256, 0, stream>>>(Wv,  wvT,  Lm, Dm,  Dm);
  cvt_t_kernel<<<cblk((size_t)Lm*Dm*Dm), 256, 0, stream>>>(Wo,  woT,  Lm, Dm,  Dm);
  cvt_t_kernel<<<cblk((size_t)Lm*Dm*FFm),256, 0, stream>>>(Wi,  wiT,  Lm, Dm,  FFm);
  cvt_t_kernel<<<cblk((size_t)Lm*Dm*FFm),256, 0, stream>>>(Wo2, wo2T, Lm, FFm, Dm);
  int nlora = Tm * Lm * Rm * Dm;
  cvt_kernel<<<cblk(nlora), 256, 0, stream>>>(qA, qA_b, nlora);
  cvt_kernel<<<cblk(nlora), 256, 0, stream>>>(qB, qB_b, nlora);
  cvt_kernel<<<cblk(nlora), 256, 0, stream>>>(vA, vA_b, nlora);
  cvt_kernel<<<cblk(nlora), 256, 0, stream>>>(vB, vB_b, nlora);

  auto gemm = [&](const bf16* A, const bf16* Bt, const float* bias,
                  float* Cf, bf16* Cb, int N, int K, int act) {
    dim3 grid(N / 16, Mm / 128);
    gemm_bf16_kernel<<<grid, 256, 0, stream>>>(A, Bt, bias, Cf, Cb, Mm, N, K, act);
  };

  auto run_pass = [&](bool lora) {
    for (int l = 0; l < Lm; ++l) {
      size_t wo_off = (size_t)l * Dm * Dm;
      size_t wi_off = (size_t)l * Dm * FFm;
      gemm(h_b, wqT + wo_off, bq + l * Dm, q_f, q_b, Dm, Dm, 0);
      gemm(h_b, wkT + wo_off, bk + l * Dm, nullptr, k_b, Dm, Dm, 0);
      gemm(h_b, wvT + wo_off, bv + l * Dm, v_f, v_b, Dm, Dm, 0);
      if (lora) {
        lora_gemm1_kernel<<<dim3(1, 16), 256, 0, stream>>>(h_b, qA_b, task, l, tmp_b);
        lora_gemm2_kernel<<<dim3(48, 16), 256, 0, stream>>>(tmp_b, qB_b, task, l, q_f, q_b);
        lora_gemm1_kernel<<<dim3(1, 16), 256, 0, stream>>>(h_b, vA_b, task, l, tmp_b);
        lora_gemm2_kernel<<<dim3(48, 16), 256, 0, stream>>>(tmp_b, vB_b, task, l, v_f, v_b);
      }
      attn_kernel<<<Bm * Hm * (Sm / 16), 32, 0, stream>>>(q_b, k_b, v_b, amask, o_b);
      gemm(o_b, woT + wo_off, bo + l * Dm, t1_f, nullptr, Dm, Dm, 0);
      add_ln_kernel<<<Mm, 256, 0, stream>>>(h_f, t1_f, g1 + l * Dm, b1 + l * Dm, h1_f, h1_b);
      gemm(h1_b, wiT + wi_off, bi + l * FFm, nullptr, ff1_b, FFm, Dm, 1);
      gemm(ff1_b, wo2T + wi_off, bo2 + l * Dm, ff2_f, nullptr, Dm, FFm, 0);
      add_ln_kernel<<<Mm, 256, 0, stream>>>(h1_f, ff2_f, g2 + l * Dm, b2 + l * Dm, h_f, h_b);
    }
  };

  // ---- pass 1 (no LoRA) ----
  embed_ln_kernel<<<Mm, 256, 0, stream>>>(ids, wemb, pemb, emb_g, emb_b, h_f, h_b);
  run_pass(false);

  // ---- router (writes out[0..23] and task[]) ----
  router_kernel<<<Bm, 256, 0, stream>>>(h_f, rW1, rb1, rW2, rb2, temp, out, task);

  // ---- pass 2 (LoRA-routed) ----
  embed_ln_kernel<<<Mm, 256, 0, stream>>>(ids, wemb, pemb, emb_g, emb_b, h_f, h_b);
  run_pass(true);

  // ---- classifier (writes out[24..39]) ----
  cls_kernel<<<Bm, 256, 0, stream>>>(h_f, cW1, cb1, cW2, cb2, task, out + Bm * Tm);
}
